// Core_70325794505291
// MI455X (gfx1250) — matrix operand order repaired and checked by reading, not compiled, not run on hardware
//
#include <hip/hip_runtime.h>

// ---------- types / helpers ----------
typedef __bf16 v16bf __attribute__((ext_vector_type(16)));
typedef float  v8f   __attribute__((ext_vector_type(8)));

struct U32x8 { uint4 lo, hi; };

static __device__ __forceinline__ v16bf frag_cast(uint4 lo, uint4 hi) {
  U32x8 t; t.lo = lo; t.hi = hi;
  return __builtin_bit_cast(v16bf, t);
}
// f32 -> bf16 (round to nearest even)
static __device__ __forceinline__ unsigned short f2bf(float f) {
  unsigned int u = __builtin_bit_cast(unsigned int, f);
  unsigned int r = u + 0x7FFFu + ((u >> 16) & 1u);
  return (unsigned short)(r >> 16);
}
static __device__ __forceinline__ float bf2f(unsigned short h) {
  unsigned int u = ((unsigned int)h) << 16;
  return __builtin_bit_cast(float, u);
}

// gfx1250 async global->LDS copy (ASYNCcnt-tracked, no VGPR round trip).
// LDS aperture places the byte offset in addr[31:0], so the low 32 bits of a
// generic pointer cast give the VDST LDS address.
static __device__ __forceinline__ void async_g2l_b128(void* lds, const void* gptr) {
  unsigned l = (unsigned)(unsigned long long)(size_t)lds;
  unsigned long long g = (unsigned long long)(size_t)gptr;
  asm volatile("global_load_async_to_lds_b128 %0, %1, off" :: "v"(l), "v"(g) : "memory");
}
static __device__ __forceinline__ void wait_async0() {
  asm volatile("s_wait_asynccnt 0x0" ::: "memory");
}

// ---------- stage 0: normalize + permute + space_to_depth -> NHWC bf16 halves ----------
__global__ void prep_kernel(const float* __restrict__ x, const float* __restrict__ mu,
                            const float* __restrict__ sigma, const int* __restrict__ perm,
                            unsigned short* __restrict__ a1, unsigned short* __restrict__ a2) {
  int idx = blockIdx.x * blockDim.x + threadIdx.x;   // 8*32*32*192
  if (idx >= 8 * 32 * 32 * 192) return;
  int ch = idx % 192; int rest = idx / 192;
  int W = rest & 31; rest >>= 5;
  int H = rest & 31; rest >>= 5;
  int b = rest;
  int c = ch >> 6, r = ch & 63, bi = r >> 3, bj = r & 7;
  int p = ((H * 8 + bi) << 8) + (W * 8 + bj);        // pixel in 256x256 plane
  int s = perm[(c << 16) + p];
  float v = (x[((b * 3 + c) << 16) + s] - mu[c]) / sigma[c];
  unsigned short hv = f2bf(v);
  int pix = (b * 32 + H) * 32 + W;
  if (ch < 96) a1[(size_t)pix * 96 + ch] = hv;
  else         a2[(size_t)pix * 96 + (ch - 96)] = hv;
}

// ---------- weight repack into WMMA B-fragment order: [kchunk][ntile][lane(32)][16] ----------
__global__ void pack_w_kernel(const float* __restrict__ w, unsigned short* __restrict__ wp, int total) {
  int idx = blockIdx.x * blockDim.x + threadIdx.x;
  if (idx >= total) return;
  const int WBLK = 737280;
  const int cum[7] = {0, 110592, 258048, 368640, 479232, 626688, 737280};
  const int OCs[6] = {128, 128, 96, 128, 128, 96};
  const int ICs[6] = {96, 128, 128, 96, 128, 128};
  int blk = idx / WBLK;
  int rem = idx - blk * WBLK;
  int j = 0;
  #pragma unroll
  for (int t = 1; t < 6; ++t) if (rem >= cum[t]) j = t;
  int li = rem - cum[j];
  int OC = OCs[j], IC = ICs[j];
  int NT = OC >> 4;
  int i    = li & 15;
  int lane = (li >> 4) & 31;
  int nt   = (li >> 9) % NT;
  int kc   = li / (NT << 9);
  int oc = nt * 16 + (lane & 15);
  int k  = kc * 32 + ((lane >> 4) << 4) + i;         // K order: (kh*3+kw)*IC + c
  int tap = k / IC, cidx = k - tap * IC;
  int kh = tap / 3, kw = tap - kh * 3;
  int woff = blk * WBLK + cum[j];
  wp[idx] = f2bf(w[(size_t)woff + ((size_t)(oc * IC + cidx) * 3 + kh) * 3 + kw]);
}

// ---------- 1x1 mix-matrix repack: [blk][kchunk(6)][ntile(12)][lane][16] ----------
__global__ void pack_m_kernel(const float* __restrict__ m, unsigned short* __restrict__ mp) {
  int idx = blockIdx.x * blockDim.x + threadIdx.x;
  if (idx >= 32 * 192 * 192) return;
  int blk = idx / 36864;
  int li = idx - blk * 36864;
  int i    = li & 15;
  int lane = (li >> 4) & 31;
  int nt   = (li >> 9) % 12;
  int kc   = li / 6144;
  int oc = nt * 16 + (lane & 15);
  int k  = kc * 32 + ((lane >> 4) << 4) + i;
  mp[idx] = f2bf(m[(size_t)blk * 36864 + oc * 192 + k]);   // B[k][n] = M[o=n][c=k]
}

// ---------- implicit-GEMM 3x3 conv via v_wmma_f32_16x16x32_bf16 ----------
// Tile: 128 output pixels (4 rows of one image) x (OC/2) columns per workgroup.
// 8 waves; each wave owns 16 pixels (M) and its N tiles as v8f accumulators.
// Weights stream through a double-buffered LDS chunk via async global->LDS,
// overlapped with the WMMA work on the other buffer (one barrier per K step).
template <int IC, int OC, bool RELU, bool RES>
__global__ __launch_bounds__(256)
void conv3x3_wmma(const unsigned short* __restrict__ in,   // NHWC (8192, IC) bf16
                  unsigned short* __restrict__ out,        // NHWC (8192, OC) bf16 (also residual src if RES)
                  const unsigned short* __restrict__ wp,   // packed [KC][OC/16][32][16]
                  const float* __restrict__ bias) {
  constexpr int KC  = IC * 9 / 32;    // K chunks of 32
  constexpr int NTG = OC / 16;        // global N tiles
  constexpr int NT  = NTG / 2;        // N tiles per workgroup (y-split)
  constexpr int BCH = (OC / 2) * 32;  // B chunk elements per workgroup
  __shared__ __align__(16) unsigned short sA[6 * 34 * IC];
  __shared__ __align__(16) unsigned short sB[2][BCH];

  const int tid = threadIdx.x, lane = tid & 31, wave = tid >> 5;
  const int tile = blockIdx.x;                 // 0..63
  const int ns = blockIdx.y;                   // 0..1
  const int n0t = ns * NT;
  const int b  = tile >> 3;
  const int h0 = (tile & 7) << 2;

  // async stage of one B chunk into buffer `buf`
  auto stageB = [&](int kc, int buf) {
    const unsigned short* src = wp + ((size_t)kc * NTG + n0t) * 512;
    constexpr int BV = BCH / 8;
    for (int v = tid; v < BV; v += 256)
      async_g2l_b128(&sB[buf][(size_t)v * 8], src + (size_t)v * 8);
  };

  // stage input tile with halo: rows h0-1..h0+4, cols -1..32, all channels
  // (in-bounds: async global->LDS; halo: zero via ds_store)
  {
    constexpr int CV = IC / 8;
    constexpr int VEC = 6 * 34 * CV;
    for (int v = tid; v < VEC; v += 256) {
      int cg  = v % CV;
      int t2  = v / CV;
      int col = t2 % 34;
      int row = t2 / 34;
      int H = h0 + row - 1, W = col - 1;
      if (H >= 0 && H < 32 && W >= 0 && W < 32)
        async_g2l_b128(sA + (size_t)v * 8,
                       in + ((size_t)((b * 32 + H) * 32 + W) * IC + cg * 8));
      else
        *reinterpret_cast<uint4*>(sA + (size_t)v * 8) = make_uint4(0u, 0u, 0u, 0u);
    }
  }
  stageB(0, 0);

  // bias folded into accumulator init: every element of a C fragment in a lane has fixed N
  v8f acc[NT];
  #pragma unroll
  for (int nt = 0; nt < NT; ++nt) {
    float bv = bias[(n0t + nt) * 16 + (lane & 15)];
    #pragma unroll
    for (int r = 0; r < 8; ++r) acc[nt][r] = bv;
  }

  const int pl = wave * 16 + (lane & 15);      // wave-local pixel (A row M)
  const int lr = pl >> 5, lw = pl & 31;
  const int chalf = (lane >> 4) << 4;          // A-layout lane split (K +0 / +16)
  constexpr int CPT = IC / 32;                 // channel chunks per tap

  wait_async0();
  __syncthreads();

  for (int kc = 0; kc < KC; ++kc) {
    if (kc + 1 < KC) stageB(kc + 1, (kc + 1) & 1);   // overlap fetch with compute
    if (kc + 2 < KC)
      __builtin_prefetch(wp + ((size_t)(kc + 2) * NTG + n0t) * 512, 0, 1);

    int tap = kc / CPT, cc = kc - tap * CPT;
    int kh = tap / 3, kw = tap - kh * 3;
    int aoff = ((lr + kh) * 34 + (lw + kw)) * IC + cc * 32 + chalf;
    uint4 alo = *reinterpret_cast<const uint4*>(sA + aoff);       // K = c..c+7
    uint4 ahi = *reinterpret_cast<const uint4*>(sA + aoff + 8);   // K = c+8..c+15
    v16bf afrag = frag_cast(alo, ahi);

    const unsigned short* sb = sB[kc & 1];
    #pragma unroll
    for (int nt = 0; nt < NT; ++nt) {
      int boff = (nt * 32 + lane) * 16;
      v16bf bfrag = frag_cast(*reinterpret_cast<const uint4*>(sb + boff),
                              *reinterpret_cast<const uint4*>(sb + boff + 8));
      acc[nt] = __builtin_amdgcn_wmma_f32_16x16x32_bf16(
          false, afrag, false, bfrag, (short)0, acc[nt], false, false);
    }

    wait_async0();          // next chunk landed in LDS
    __syncthreads();        // publish to all waves / allow next overwrite
  }

  // epilogue: relu / residual-add / store bf16 NHWC
  const int Mb = (lane >> 4) << 3;
  #pragma unroll
  for (int nt = 0; nt < NT; ++nt) {
    int n = (n0t + nt) * 16 + (lane & 15);
    #pragma unroll
    for (int r = 0; r < 8; ++r) {
      float v = acc[nt][r];
      if (RELU) v = v > 0.f ? v : 0.f;
      int pixel = tile * 128 + wave * 16 + Mb + r;
      size_t o = (size_t)pixel * OC + n;
      if (RES) v += bf2f(out[o]);
      out[o] = f2bf(v);
    }
  }
}

// ---------- 192x192 1x1 mix (einsum) ----------
template <bool FINAL>
__global__ __launch_bounds__(256)
void mix1x1_wmma(const unsigned short* __restrict__ in1, const unsigned short* __restrict__ in2,
                 const unsigned short* __restrict__ mp,   // packed [6][12][32][16]
                 unsigned short* __restrict__ out1, unsigned short* __restrict__ out2,
                 float* __restrict__ outf) {
  __shared__ __align__(16) unsigned short sA[128 * 192];
  __shared__ __align__(16) unsigned short sB[2][96 * 32];
  const int tid = threadIdx.x, lane = tid & 31, wave = tid >> 5;
  const int tile = blockIdx.x;                 // 0..63, 128 pixels each
  const int ns = blockIdx.y;                   // 0..1 -> N cols [ns*96, ns*96+96)
  const int n0t = ns * 6;

  auto stageB = [&](int kc, int buf) {
    const unsigned short* src = mp + ((size_t)kc * 12 + n0t) * 512;
    for (int v = tid; v < 96 * 32 / 8; v += 256)
      async_g2l_b128(&sB[buf][(size_t)v * 8], src + (size_t)v * 8);
  };

  // stage concat(A1,A2) as K=192 contiguous channels per pixel (all async)
  for (int v = tid; v < 128 * 192 / 8; v += 256) {
    int cg = v % 24;
    int p  = v / 24;
    int pixel = tile * 128 + p;
    const unsigned short* src = (cg < 12)
        ? in1 + (size_t)pixel * 96 + cg * 8
        : in2 + (size_t)pixel * 96 + (cg - 12) * 8;
    async_g2l_b128(sA + (size_t)v * 8, src);
  }
  stageB(0, 0);

  v8f acc[6];
  #pragma unroll
  for (int nt = 0; nt < 6; ++nt)
    #pragma unroll
    for (int r = 0; r < 8; ++r) acc[nt][r] = 0.f;

  const int pl = wave * 16 + (lane & 15);
  const int chalf = (lane >> 4) << 4;
  wait_async0();
  __syncthreads();

  for (int kc = 0; kc < 6; ++kc) {
    if (kc + 1 < 6) stageB(kc + 1, (kc + 1) & 1);

    int aoff = pl * 192 + kc * 32 + chalf;
    v16bf afrag = frag_cast(*reinterpret_cast<const uint4*>(sA + aoff),
                            *reinterpret_cast<const uint4*>(sA + aoff + 8));
    const unsigned short* sb = sB[kc & 1];
    #pragma unroll
    for (int nt = 0; nt < 6; ++nt) {
      int boff = (nt * 32 + lane) * 16;
      v16bf bfrag = frag_cast(*reinterpret_cast<const uint4*>(sb + boff),
                              *reinterpret_cast<const uint4*>(sb + boff + 8));
      acc[nt] = __builtin_amdgcn_wmma_f32_16x16x32_bf16(
          false, afrag, false, bfrag, (short)0, acc[nt], false, false);
    }

    wait_async0();
    __syncthreads();
  }

  const int Mb = (lane >> 4) << 3;
  #pragma unroll
  for (int nt = 0; nt < 6; ++nt) {
    int n = (n0t + nt) * 16 + (lane & 15);     // 0..191
    #pragma unroll
    for (int r = 0; r < 8; ++r) {
      float v = acc[nt][r];
      int pixel = tile * 128 + wave * 16 + Mb + r;
      if (FINAL) {
        int bimg = pixel >> 10, hw = pixel & 1023;
        outf[((size_t)bimg * 192 + n) * 1024 + hw] = v;   // NCHW f32
      } else {
        if (n < 96) out1[(size_t)pixel * 96 + n]        = f2bf(v);
        else        out2[(size_t)pixel * 96 + (n - 96)] = f2bf(v);
      }
    }
  }
}

// ---------- host orchestration ----------
extern "C" void kernel_launch(void* const* d_in, const int* in_sizes, int n_in,
                              void* d_out, int out_size, void* d_ws, size_t ws_size,
                              hipStream_t stream) {
  (void)in_sizes; (void)n_in; (void)out_size; (void)ws_size;
  const float* x      = (const float*)d_in[0];
  const float* mu     = (const float*)d_in[1];
  const float* sigma  = (const float*)d_in[2];
  const float* w_flat = (const float*)d_in[3];
  const float* b_flat = (const float*)d_in[4];
  const float* m      = (const float*)d_in[5];
  const int*   perm   = (const int*)d_in[6];

  char* ws = (char*)d_ws;
  const size_t S96  = (size_t)8 * 32 * 32 * 96 * 2;    // 1.5 MB bf16 half-tensor
  const size_t S128 = (size_t)8 * 32 * 32 * 128 * 2;
  const int W_TOTAL_ = 23592960;
  unsigned short* A1 = (unsigned short*)(ws);
  unsigned short* A2 = (unsigned short*)(ws + S96);
  unsigned short* B1 = (unsigned short*)(ws + 2 * S96);
  unsigned short* B2 = (unsigned short*)(ws + 3 * S96);
  unsigned short* T1 = (unsigned short*)(ws + 4 * S96);
  unsigned short* T2 = (unsigned short*)(ws + 4 * S96 + S128);
  unsigned short* WP = (unsigned short*)(ws + 4 * S96 + 2 * S128);
  unsigned short* MP = (unsigned short*)(ws + 4 * S96 + 2 * S128 + (size_t)W_TOTAL_ * 2);

  prep_kernel<<<(8 * 32 * 32 * 192 + 255) / 256, 256, 0, stream>>>(x, mu, sigma, perm, A1, A2);
  pack_w_kernel<<<(W_TOTAL_ + 255) / 256, 256, 0, stream>>>(w_flat, WP, W_TOTAL_);
  pack_m_kernel<<<(32 * 192 * 192 + 255) / 256, 256, 0, stream>>>(m, MP);

  const int cumW[6] = {0, 110592, 258048, 368640, 479232, 626688};
  const int cumB[6] = {0, 128, 256, 352, 480, 608};
  unsigned short* cur1 = A1; unsigned short* cur2 = A2;
  unsigned short* nxt1 = B1; unsigned short* nxt2 = B2;
  dim3 grid(64, 2), blk(256);

  for (int b = 0; b < 32; ++b) {
    for (int rnd = 0; rnd < 2; ++rnd) {
      int j0 = rnd * 3;
      const unsigned short* w0 = WP + (size_t)b * 737280 + cumW[j0 + 0];
      const unsigned short* w1 = WP + (size_t)b * 737280 + cumW[j0 + 1];
      const unsigned short* w2 = WP + (size_t)b * 737280 + cumW[j0 + 2];
      const float* bb0 = b_flat + (size_t)b * 704 + cumB[j0 + 0];
      const float* bb1 = b_flat + (size_t)b * 704 + cumB[j0 + 1];
      const float* bb2 = b_flat + (size_t)b * 704 + cumB[j0 + 2];
      conv3x3_wmma< 96, 128, true,  false><<<grid, blk, 0, stream>>>(cur2, T1,   w0, bb0);
      conv3x3_wmma<128, 128, true,  false><<<grid, blk, 0, stream>>>(T1,   T2,   w1, bb1);
      conv3x3_wmma<128,  96, false, true ><<<grid, blk, 0, stream>>>(T2,   cur1, w2, bb2);
      unsigned short* t = cur1; cur1 = cur2; cur2 = t;   // op9 swap
    }
    const unsigned short* mpp = MP + (size_t)b * 36864;
    if (b == 31) {
      mix1x1_wmma<true ><<<grid, blk, 0, stream>>>(cur1, cur2, mpp, nullptr, nullptr, (float*)d_out);
    } else {
      mix1x1_wmma<false><<<grid, blk, 0, stream>>>(cur1, cur2, mpp, nxt1, nxt2, nullptr);
      unsigned short* t1 = cur1; unsigned short* t2 = cur2;
      cur1 = nxt1; cur2 = nxt2; nxt1 = t1; nxt2 = t2;    // ping-pong reg0 buffers
    }
  }
}
